// GAT_MINI_DGL_38225208934549
// MI455X (gfx1250) — matrix operand order repaired and checked by reading, not compiled, not run on hardware
//
#include <hip/hip_runtime.h>
#include <hip/hip_bf16.h>
#include <hip/hip_fp16.h>

typedef __attribute__((ext_vector_type(16))) _Float16 v16h;
typedef __attribute__((ext_vector_type(8)))  float    v8f;

#define N_NODES   100000
#define N_EDGES   1600000
#define IN_DIM    256
#define HIDDEN    64
#define HEADS     8
#define DHEAD     8        // per-head dim, layer 1
#define N_CLASSES 47
#define L2_STRIDE 48       // padded column stride for layer-2 features
#define NEG_SLOPE 0.2f

// ---------------------------------------------------------------- utilities

__device__ __forceinline__ float lrelu(float x) {
    return x > 0.0f ? x : NEG_SLOPE * x;
}

// float atomic max via int/uint bit trick (works for all sign combinations)
__device__ __forceinline__ void atomicMaxF(float* addr, float val) {
    if (val >= 0.0f)
        atomicMax(reinterpret_cast<int*>(addr), __float_as_int(val));
    else
        atomicMin(reinterpret_cast<unsigned int*>(addr), __float_as_uint(val));
}

__global__ void fill_f32(float* __restrict__ p, long long n, float v) {
    long long i = (long long)blockIdx.x * blockDim.x + threadIdx.x;
    if (i < n) p[i] = v;
}

// Transpose + convert weights: W[K,N] f32 row-major -> Wt[Npad,K] f16,
// rows n >= N are zero-filled (so padded output columns are exactly zero).
__global__ void weights_to_f16t(const float* __restrict__ W, int K, int N, int Npad,
                                _Float16* __restrict__ Wt) {
    long long i = (long long)blockIdx.x * blockDim.x + threadIdx.x;
    if (i >= (long long)Npad * K) return;
    int n = (int)(i / K);
    int k = (int)(i % K);
    Wt[i] = (n < N) ? (_Float16)W[(size_t)k * N + n] : (_Float16)0.0f;
}

// ---------------------------------------------------------------- WMMA GEMM
// Dual GEMM: one wave computes a 16-row slab times ALL columns of BOTH
// projections:  C0 = A @ W0t^T + bias0,  C1 = A @ W1t^T + bias1.
// A is f32 [M,KK] (converted to f16 on the fly, read exactly once);
// Wt are f16 [NCOLT*16, KK] transposed+padded so a B fragment is one
// contiguous 32-byte per-lane load.  KK % 32 == 0, M % 16 == 0.
template <int NCOLT, int KK>
__global__ __launch_bounds__(32)
void wmma_dual_gemm(const float* __restrict__ A, int lda,
                    const _Float16* __restrict__ Wt0,
                    const _Float16* __restrict__ Wt1,
                    const float* __restrict__ bias0,
                    const float* __restrict__ bias1,
                    float* __restrict__ C0, float* __restrict__ C1,
                    int ldc, int ncols /*valid cols for bias*/) {
    const int lane = threadIdx.x;          // 0..31
    const int row0 = blockIdx.x * 16;
    const int mrow = lane & 15;
    const int hi   = lane >> 4;            // 0 or 1

    v8f acc[2][NCOLT];
#pragma unroll
    for (int m = 0; m < 2; ++m)
#pragma unroll
        for (int c = 0; c < NCOLT; ++c)
            acc[m][c] = (v8f){};

    const float* Arow = A + (size_t)(row0 + mrow) * lda;
    const _Float16* Wts[2] = {Wt0, Wt1};

#pragma unroll 1
    for (int k0 = 0; k0 < KK; k0 += 32) {
        // ---- A fragment: 16x32 f16. lane holds row M=lane&15;
        //      half h -> k = (lane>=16?16:0) + h
        v16h a;
        {
            const float4* p0 = reinterpret_cast<const float4*>(Arow + k0 + hi * 16);
            const float4* p1 = reinterpret_cast<const float4*>(Arow + k0 + hi * 16 + 8);
            float4 r0 = p0[0], r1 = p0[1];
            float4 r2 = p1[0], r3 = p1[1];
            a[0]  = (_Float16)r0.x; a[1]  = (_Float16)r0.y;
            a[2]  = (_Float16)r0.z; a[3]  = (_Float16)r0.w;
            a[4]  = (_Float16)r1.x; a[5]  = (_Float16)r1.y;
            a[6]  = (_Float16)r1.z; a[7]  = (_Float16)r1.w;
            a[8]  = (_Float16)r2.x; a[9]  = (_Float16)r2.y;
            a[10] = (_Float16)r2.z; a[11] = (_Float16)r2.w;
            a[12] = (_Float16)r3.x; a[13] = (_Float16)r3.y;
            a[14] = (_Float16)r3.z; a[15] = (_Float16)r3.w;
        }
        // ---- B fragments: lane holds col N=col0+(lane&15), halves are
        //      k = k0 + (lane>=16?16:0) + h -> contiguous in Wt row.
#pragma unroll
        for (int m = 0; m < 2; ++m) {
#pragma unroll
            for (int c = 0; c < NCOLT; ++c) {
                const int n = c * 16 + mrow;
                const _Float16* bp = Wts[m] + (size_t)n * KK + k0 + hi * 16;
                v16h b = *reinterpret_cast<const v16h*>(bp);
                acc[m][c] = __builtin_amdgcn_wmma_f32_16x16x32_f16(
                    false, a, false, b, (short)0, acc[m][c], false, false);
            }
        }
    }

    // ---- epilogue: VGPR v holds row (row0 + v + hi*8), col (c*16 + lane&15)
    float* Cs[2] = {C0, C1};
    const float* Bs[2] = {bias0, bias1};
#pragma unroll
    for (int m = 0; m < 2; ++m) {
#pragma unroll
        for (int c = 0; c < NCOLT; ++c) {
            const int ncol = c * 16 + mrow;
            const float bv = Bs[m][ncol < ncols ? ncol : 0];  // pad col value unused
#pragma unroll
            for (int v = 0; v < 8; ++v)
                Cs[m][(size_t)(row0 + hi * 8 + v) * ldc + ncol] = acc[m][c][v] + bv;
        }
    }
}

// ------------------------------------------------------- layer-1 edge passes
// thread = edge*8 + head
__global__ void edge_score1(const float* __restrict__ fs1,
                            const float* __restrict__ fd1,
                            const long long* __restrict__ src,
                            const long long* __restrict__ dst,
                            const float* __restrict__ attn1,
                            float* __restrict__ e1, float* __restrict__ m1) {
    long long t = (long long)blockIdx.x * blockDim.x + threadIdx.x;
    if (t >= (long long)N_EDGES * HEADS) return;
    long long e = t >> 3;
    int h = (int)(t & 7);
    long long u = src[e], v = dst[e];
    const float4* ps = reinterpret_cast<const float4*>(fs1 + u * HIDDEN + h * DHEAD);
    const float4* pd = reinterpret_cast<const float4*>(fd1 + v * HIDDEN + h * DHEAD);
    float4 s0 = ps[0], s1v = ps[1], d0 = pd[0], d1 = pd[1];
    const float* aw = attn1 + h * DHEAD;
    float sum = aw[0] * lrelu(s0.x + d0.x) + aw[1] * lrelu(s0.y + d0.y)
              + aw[2] * lrelu(s0.z + d0.z) + aw[3] * lrelu(s0.w + d0.w)
              + aw[4] * lrelu(s1v.x + d1.x) + aw[5] * lrelu(s1v.y + d1.y)
              + aw[6] * lrelu(s1v.z + d1.z) + aw[7] * lrelu(s1v.w + d1.w);
    e1[t] = sum;
    atomicMaxF(&m1[v * HEADS + h], sum);
}

__global__ void edge_exp1(const long long* __restrict__ dst,
                          const float* __restrict__ m1,
                          float* __restrict__ e1, float* __restrict__ s1) {
    long long t = (long long)blockIdx.x * blockDim.x + threadIdx.x;
    if (t >= (long long)N_EDGES * HEADS) return;
    long long e = t >> 3;
    int h = (int)(t & 7);
    long long v = dst[e];
    float ex = __expf(e1[t] - m1[v * HEADS + h]);
    e1[t] = ex;
    atomicAdd(&s1[v * HEADS + h], ex);
}

__global__ void edge_agg1(const float* __restrict__ fs1,
                          const long long* __restrict__ src,
                          const long long* __restrict__ dst,
                          const float* __restrict__ e1,
                          const float* __restrict__ s1,
                          float* __restrict__ agg) {
    long long t = (long long)blockIdx.x * blockDim.x + threadIdx.x;
    if (t >= (long long)N_EDGES * HEADS) return;
    long long e = t >> 3;
    int h = (int)(t & 7);
    __builtin_prefetch(&src[e + 8192], 0, 0);  // speculative; OOB is dropped
    long long u = src[e], v = dst[e];
    float sv = s1[v * HEADS + h];
    sv = (sv == 0.0f) ? 1.0f : sv;
    float a = e1[t] / sv;
    const float* fsp = fs1 + u * HIDDEN + h * DHEAD;
    float* op = agg + v * HIDDEN + h * DHEAD;
#pragma unroll
    for (int d = 0; d < DHEAD; ++d)
        atomicAdd(&op[d], a * fsp[d]);
}

__global__ void node_elu1(float* __restrict__ agg, const float* __restrict__ bias1) {
    long long t = (long long)blockIdx.x * blockDim.x + threadIdx.x;
    if (t >= (long long)N_NODES * HIDDEN) return;
    float x = agg[t] + bias1[t % HIDDEN];
    agg[t] = x > 0.0f ? x : expm1f(x);
}

// ------------------------------------------------------- layer-2 edge passes
// thread = edge (single head)
__global__ void edge_score2(const float* __restrict__ fs2,
                            const float* __restrict__ fd2,
                            const long long* __restrict__ src,
                            const long long* __restrict__ dst,
                            const float* __restrict__ attn2,
                            float* __restrict__ e2, float* __restrict__ m2) {
    long long t = (long long)blockIdx.x * blockDim.x + threadIdx.x;
    if (t >= (long long)N_EDGES) return;
    long long u = src[t], v = dst[t];
    const float* ps = fs2 + u * L2_STRIDE;
    const float* pd = fd2 + v * L2_STRIDE;
    float sum = 0.0f;
#pragma unroll 1
    for (int c = 0; c < N_CLASSES; ++c)
        sum += attn2[c] * lrelu(ps[c] + pd[c]);
    e2[t] = sum;
    atomicMaxF(&m2[v], sum);
}

__global__ void edge_exp2(const long long* __restrict__ dst,
                          const float* __restrict__ m2,
                          float* __restrict__ e2, float* __restrict__ s2) {
    long long t = (long long)blockIdx.x * blockDim.x + threadIdx.x;
    if (t >= (long long)N_EDGES) return;
    long long v = dst[t];
    float ex = __expf(e2[t] - m2[v]);
    e2[t] = ex;
    atomicAdd(&s2[v], ex);
}

__global__ void edge_agg2(const float* __restrict__ fs2,
                          const long long* __restrict__ src,
                          const long long* __restrict__ dst,
                          const float* __restrict__ e2,
                          const float* __restrict__ s2,
                          float* __restrict__ out) {
    long long t = (long long)blockIdx.x * blockDim.x + threadIdx.x;
    if (t >= (long long)N_EDGES) return;
    __builtin_prefetch(&src[t + 8192], 0, 0);  // speculative; OOB is dropped
    long long u = src[t], v = dst[t];
    float sv = s2[v];
    sv = (sv == 0.0f) ? 1.0f : sv;
    float a = e2[t] / sv;
    const float* fsp = fs2 + u * L2_STRIDE;
    float* op = out + v * N_CLASSES;
#pragma unroll 1
    for (int c = 0; c < N_CLASSES; ++c)
        atomicAdd(&op[c], a * fsp[c]);
}

__global__ void node_bias2(float* __restrict__ out, const float* __restrict__ bias2) {
    long long t = (long long)blockIdx.x * blockDim.x + threadIdx.x;
    if (t >= (long long)N_NODES * N_CLASSES) return;
    out[t] += bias2[t % N_CLASSES];
}

// ---------------------------------------------------------------- launcher
extern "C" void kernel_launch(void* const* d_in, const int* in_sizes, int n_in,
                              void* d_out, int out_size, void* d_ws, size_t ws_size,
                              hipStream_t stream) {
    const float*     x      = (const float*)d_in[0];
    const long long* src    = (const long long*)d_in[1];
    const long long* dst    = (const long long*)d_in[2];
    const float*     W1s    = (const float*)d_in[3];
    const float*     b1s    = (const float*)d_in[4];
    const float*     W1d    = (const float*)d_in[5];
    const float*     b1d    = (const float*)d_in[6];
    const float*     attn1  = (const float*)d_in[7];
    const float*     bias1  = (const float*)d_in[8];
    const float*     W2s    = (const float*)d_in[9];
    const float*     b2s    = (const float*)d_in[10];
    const float*     W2d    = (const float*)d_in[11];
    const float*     b2d    = (const float*)d_in[12];
    const float*     attn2  = (const float*)d_in[13];
    const float*     bias2  = (const float*)d_in[14];
    float*           out    = (float*)d_out;

    // workspace layout (floats)
    float* ws  = (float*)d_ws;
    float* fs1 = ws;                                   // N*64
    float* fd1 = fs1 + (size_t)N_NODES * HIDDEN;       // N*64
    float* h1  = fd1 + (size_t)N_NODES * HIDDEN;       // N*64 (agg1, then elu'd)
    float* m1  = h1  + (size_t)N_NODES * HIDDEN;       // N*8
    float* s1  = m1  + (size_t)N_NODES * HEADS;        // N*8
    float* e1  = s1  + (size_t)N_NODES * HEADS;        // E*8
    // f16 transposed weights after the big regions (32B-aligned by construction)
    _Float16* wt1s = (_Float16*)(e1 + (size_t)N_EDGES * HEADS);      // 64*256
    _Float16* wt1d = wt1s + (size_t)HIDDEN * IN_DIM;                 // 64*256
    _Float16* wt2s = wt1d + (size_t)HIDDEN * IN_DIM;                 // 48*64
    _Float16* wt2d = wt2s + (size_t)L2_STRIDE * HIDDEN;              // 48*64
    // layer-2 reuse of dead layer-1 regions
    float* fs2 = fs1;   // N*48 <= N*64
    float* fd2 = fd1;
    float* m2  = m1;    // N
    float* s2  = s1;
    float* e2  = e1;    // E

    const float NEG_INF = -__builtin_huge_valf();
    const int TB = 256;
    auto blocks = [](long long n, int tb) { return (unsigned)((n + tb - 1) / tb); };

    // ---- init layer-1 reductions
    fill_f32<<<blocks((long long)N_NODES * HEADS, TB), TB, 0, stream>>>(m1, (long long)N_NODES * HEADS, NEG_INF);
    fill_f32<<<blocks((long long)N_NODES * HEADS, TB), TB, 0, stream>>>(s1, (long long)N_NODES * HEADS, 0.0f);
    fill_f32<<<blocks((long long)N_NODES * HIDDEN, TB), TB, 0, stream>>>(h1, (long long)N_NODES * HIDDEN, 0.0f);

    // ---- weight pre-transpose/convert to f16 (tiny)
    weights_to_f16t<<<blocks((long long)HIDDEN * IN_DIM, TB), TB, 0, stream>>>(W1s, IN_DIM, HIDDEN, HIDDEN, wt1s);
    weights_to_f16t<<<blocks((long long)HIDDEN * IN_DIM, TB), TB, 0, stream>>>(W1d, IN_DIM, HIDDEN, HIDDEN, wt1d);
    weights_to_f16t<<<blocks((long long)L2_STRIDE * HIDDEN, TB), TB, 0, stream>>>(W2s, HIDDEN, N_CLASSES, L2_STRIDE, wt2s);
    weights_to_f16t<<<blocks((long long)L2_STRIDE * HIDDEN, TB), TB, 0, stream>>>(W2d, HIDDEN, N_CLASSES, L2_STRIDE, wt2d);

    // ---- layer-1 projections: [100000,256] @ [256,64] x2, fused (WMMA)
    wmma_dual_gemm<HIDDEN / 16, IN_DIM><<<N_NODES / 16, 32, 0, stream>>>(
        x, IN_DIM, wt1s, wt1d, b1s, b1d, fs1, fd1, HIDDEN, HIDDEN);

    // ---- layer-1 edge softmax + aggregation
    const long long EH = (long long)N_EDGES * HEADS;
    edge_score1<<<blocks(EH, TB), TB, 0, stream>>>(fs1, fd1, src, dst, attn1, e1, m1);
    edge_exp1  <<<blocks(EH, TB), TB, 0, stream>>>(dst, m1, e1, s1);
    edge_agg1  <<<blocks(EH, TB), TB, 0, stream>>>(fs1, src, dst, e1, s1, h1);
    node_elu1  <<<blocks((long long)N_NODES * HIDDEN, TB), TB, 0, stream>>>(h1, bias1);

    // ---- init layer-2 reductions + output
    fill_f32<<<blocks((long long)N_NODES, TB), TB, 0, stream>>>(m2, (long long)N_NODES, NEG_INF);
    fill_f32<<<blocks((long long)N_NODES, TB), TB, 0, stream>>>(s2, (long long)N_NODES, 0.0f);
    fill_f32<<<blocks((long long)N_NODES * N_CLASSES, TB), TB, 0, stream>>>(out, (long long)N_NODES * N_CLASSES, 0.0f);

    // ---- layer-2 projections: [100000,64] @ [64,47] x2, fused (WMMA, 48-pad)
    wmma_dual_gemm<L2_STRIDE / 16, HIDDEN><<<N_NODES / 16, 32, 0, stream>>>(
        h1, HIDDEN, wt2s, wt2d, b2s, b2d, fs2, fd2, L2_STRIDE, N_CLASSES);

    // ---- layer-2 edge softmax + aggregation
    edge_score2<<<blocks((long long)N_EDGES, TB), TB, 0, stream>>>(fs2, fd2, src, dst, attn2, e2, m2);
    edge_exp2  <<<blocks((long long)N_EDGES, TB), TB, 0, stream>>>(dst, m2, e2, s2);
    edge_agg2  <<<blocks((long long)N_EDGES, TB), TB, 0, stream>>>(fs2, src, dst, e2, s2, out);
    node_bias2 <<<blocks((long long)N_NODES * N_CLASSES, TB), TB, 0, stream>>>(out, bias2);
}
